// WindowAttention_73761768341788
// MI455X (gfx1250) — compile-verified, matrix operand order checked by reading, not run on hardware
//
#include <hip/hip_runtime.h>
#include <hip/hip_bf16.h>

typedef _Float16 v16h __attribute__((ext_vector_type(16)));
typedef _Float16 v8h  __attribute__((ext_vector_type(8)));
typedef _Float16 v4h  __attribute__((ext_vector_type(4)));
typedef float    v8f  __attribute__((ext_vector_type(8)));

#define NTOK 49
#define DIM  128
#define NH   4
#define HD   32

// ---- LDS layout (units: _Float16 halves), all offsets multiple of 8 halves ----
#define XS_OFF   0            // x tile:      [64][136]
#define XS_STR   136
#define QS_OFF   8704         // q:           [4 heads][64 tok][40]
#define KS_OFF   18944        // k:           [4 heads][64 tok][40]
#define QK_STR   40
#define QK_HEAD  2560
#define VT_OFF   29184        // v transposed:[4 heads][32 d][72]
#define VT_STR   72
#define VT_HEAD  2304
#define PS_OFF   38400        // softmax P:   [4 heads][64 tok][72]
#define PS_STR   72
#define PS_HEAD  4608
#define BM_OFF   56832        // bias+mask:   [4 heads][64][72] (cols 0..63 used)
#define BM_STR   72
#define BM_HEAD  4608
#define SMEM_HALVES 75264     // 150,528 bytes -> 2 workgroups / WGP

// Load one 16x32 (A) or 32x16 (B) f16 fragment: per-lane two 16B loads at
// half-offsets {0..7} and {16..23}; caller supplies base = &buf[row*stride + g*8].
__device__ __forceinline__ v16h ldfrag(const _Float16* p) {
  union { v16h v; v8h h[2]; } u;
  u.h[0] = *(const v8h*)(p);
  u.h[1] = *(const v8h*)(p + 16);
  return u.v;
}

__device__ __forceinline__ v8f wmma_f16(v16h a, v16h b, v8f c) {
  return __builtin_amdgcn_wmma_f32_16x16x32_f16(false, a, false, b, (short)0, c,
                                                false, false);
}

__global__ void cvt_w_f16(const float* __restrict__ w, _Float16* __restrict__ w16, int n) {
  int i = blockIdx.x * 256 + threadIdx.x;
  if (i < n) w16[i] = (_Float16)w[i];
}

__global__ __launch_bounds__(256)
void swin_attn_kernel(const float* __restrict__ x,
                      const float* __restrict__ mask,
                      const float* __restrict__ qkv_b,
                      const float* __restrict__ rel_table,
                      const _Float16* __restrict__ W16,
                      float* __restrict__ out) {
  __shared__ __align__(16) _Float16 smem[SMEM_HALVES];

  const int tid  = threadIdx.x;
  const int lane = tid & 31;
  const int wid  = tid >> 5;     // 8 waves
  const int l16  = lane & 15;
  const int g    = lane >> 4;    // lane group (K-split for A/B, row-split for C/D)
  const int b    = blockIdx.x;   // window index
  const int w    = b & 63;       // mask window id (b % NW)

  v8f vzero;
#pragma unroll
  for (int i = 0; i < 8; ++i) vzero[i] = 0.f;

  // -------- Phase 1: stage x as f16 (zero-padded 64 rows) + bias/mask table --------
  {
    const float4* x4 = (const float4*)(x + (size_t)b * NTOK * DIM);
    for (int i = tid; i < 64 * 32; i += 256) {       // 64 rows x 32 float4
      int m = i >> 5, c4 = i & 31;
      float4 val;
      if (m < NTOK) val = x4[m * 32 + c4];
      else          val = float4{0.f, 0.f, 0.f, 0.f};
      v4h hv;
      hv[0] = (_Float16)val.x; hv[1] = (_Float16)val.y;
      hv[2] = (_Float16)val.z; hv[3] = (_Float16)val.w;
      *(v4h*)&smem[XS_OFF + m * XS_STR + c4 * 4] = hv;
    }
  }
  for (int i = tid; i < NH * 64 * 64; i += 256) {
    int h = i >> 12;
    int n = (i >> 6) & 63;
    int m = i & 63;
    float val;
    if (m >= NTOK) {
      val = -30000.f;                        // kill padded key columns in softmax
    } else if (n >= NTOK) {
      val = 0.f;                             // padded query rows: never stored
    } else {
      int idx = (n / 7 - m / 7 + 6) * 13 + (n % 7 - m % 7 + 6);
      val = rel_table[idx * NH + h] + mask[((size_t)w * NTOK + n) * NTOK + m];
    }
    smem[BM_OFF + h * BM_HEAD + n * BM_STR + m] = (_Float16)val;
  }
  __syncthreads();

  const int h = wid >> 1;   // head handled by this wave
  const int u = wid & 1;    // half-split within head

  // -------- Phase 2: QKV projection (wave handles channels 96h+48u .. +47) --------
  // Hoist all 16 x-fragments once: reused across 3 n-tiles and 4 k-steps.
  v16h xfrag[4][4];   // [mt][kk]
  {
    const _Float16* xb = &smem[XS_OFF + l16 * XS_STR + g * 8];
#pragma unroll
    for (int mt = 0; mt < 4; ++mt)
#pragma unroll
      for (int kk = 0; kk < 4; ++kk)
        xfrag[mt][kk] = ldfrag(xb + mt * 16 * XS_STR + kk * 32);
  }

#pragma unroll
  for (int nt = 0; nt < 3; ++nt) {
    int ob = 96 * h + 48 * u + nt * 16 + l16;      // this lane's output channel
    float bias = qkv_b[ob];
    int s   = ob % 3;                               // 0=q 1=k 2=v
    int dch = (ob / 3) & (HD - 1);                  // head-dim index

    v8f acc[4];
#pragma unroll
    for (int mt = 0; mt < 4; ++mt) acc[mt] = vzero;

    const _Float16* wb = W16 + ob * DIM + g * 8;   // B: W^T[k][o], L2-resident
    v16h bf[4];
#pragma unroll
    for (int kk = 0; kk < 4; ++kk) bf[kk] = ldfrag(wb + kk * 32);

#pragma unroll
    for (int kk = 0; kk < 4; ++kk)
#pragma unroll
      for (int mt = 0; mt < 4; ++mt)
        acc[mt] = wmma_f16(xfrag[mt][kk], bf[kk], acc[mt]);

    // epilogue: scatter +bias to q/k row-major, v transposed
    _Float16* dst;
    int step;
    if (s == 0)      { dst = &smem[QS_OFF + h * QK_HEAD + dch];          step = QK_STR; }
    else if (s == 1) { dst = &smem[KS_OFF + h * QK_HEAD + dch];          step = QK_STR; }
    else             { dst = &smem[VT_OFF + h * VT_HEAD + dch * VT_STR]; step = 1; }
#pragma unroll
    for (int mt = 0; mt < 4; ++mt) {
#pragma unroll
      for (int r = 0; r < 8; ++r) {
        int m = mt * 16 + g * 8 + r;
        dst[m * step] = (_Float16)(acc[mt][r] + bias);
      }
    }
  }
  __syncthreads();

  // -------- Phase 3: attention for head h, query tiles {2u, 2u+1} --------
  const int qt0 = 2 * u;
  v16h aq0 = ldfrag(&smem[QS_OFF + h * QK_HEAD + (qt0 * 16 + l16) * QK_STR + g * 8]);
  v16h aq1 = ldfrag(&smem[QS_OFF + h * QK_HEAD + ((qt0 + 1) * 16 + l16) * QK_STR + g * 8]);

  v8f sacc[2][4];
#pragma unroll
  for (int qi = 0; qi < 2; ++qi)
#pragma unroll
    for (int kt = 0; kt < 4; ++kt) sacc[qi][kt] = vzero;

#pragma unroll
  for (int kt = 0; kt < 4; ++kt) {
    v16h bk = ldfrag(&smem[KS_OFF + h * QK_HEAD + (kt * 16 + l16) * QK_STR + g * 8]);
    sacc[0][kt] = wmma_f16(aq0, bk, sacc[0][kt]);
    sacc[1][kt] = wmma_f16(aq1, bk, sacc[1][kt]);
  }

  // softmax: row n lives in one 16-lane group, columns across 4 accs + 16 lanes
  const float scale = 0.17677669529663687f;  // 1/sqrt(32)
#pragma unroll
  for (int qi = 0; qi < 2; ++qi) {
    int nbase = (qt0 + qi) * 16 + g * 8;
#pragma unroll
    for (int r = 0; r < 8; ++r) {
      int n = nbase + r;
      const _Float16* bmrow = &smem[BM_OFF + h * BM_HEAD + n * BM_STR + l16];
      float v0 = sacc[qi][0][r] * scale + (float)bmrow[0];
      float v1 = sacc[qi][1][r] * scale + (float)bmrow[16];
      float v2 = sacc[qi][2][r] * scale + (float)bmrow[32];
      float v3 = sacc[qi][3][r] * scale + (float)bmrow[48];
      float mx = fmaxf(fmaxf(v0, v1), fmaxf(v2, v3));
      mx = fmaxf(mx, __shfl_xor(mx, 1, 16));
      mx = fmaxf(mx, __shfl_xor(mx, 2, 16));
      mx = fmaxf(mx, __shfl_xor(mx, 4, 16));
      mx = fmaxf(mx, __shfl_xor(mx, 8, 16));
      float e0 = __expf(v0 - mx), e1 = __expf(v1 - mx);
      float e2 = __expf(v2 - mx), e3 = __expf(v3 - mx);
      float sm = e0 + e1 + e2 + e3;
      sm += __shfl_xor(sm, 1, 16);
      sm += __shfl_xor(sm, 2, 16);
      sm += __shfl_xor(sm, 4, 16);
      sm += __shfl_xor(sm, 8, 16);
      float inv = 1.0f / sm;
      _Float16* prow = &smem[PS_OFF + h * PS_HEAD + n * PS_STR + l16];
      prow[0]  = (_Float16)(e0 * inv);
      prow[16] = (_Float16)(e1 * inv);
      prow[32] = (_Float16)(e2 * inv);
      prow[48] = (_Float16)(e3 * inv);
    }
  }

  // P(64x64) @ V(64x32): B from vT so K-pairs are contiguous
  v8f oacc[2][2];
#pragma unroll
  for (int qi = 0; qi < 2; ++qi)
#pragma unroll
    for (int dt = 0; dt < 2; ++dt) oacc[qi][dt] = vzero;

#pragma unroll
  for (int kk = 0; kk < 2; ++kk) {
    v16h bv0 = ldfrag(&smem[VT_OFF + h * VT_HEAD + l16 * VT_STR + kk * 32 + g * 8]);
    v16h bv1 = ldfrag(&smem[VT_OFF + h * VT_HEAD + (16 + l16) * VT_STR + kk * 32 + g * 8]);
    v16h ap0 = ldfrag(&smem[PS_OFF + h * PS_HEAD + (qt0 * 16 + l16) * PS_STR + kk * 32 + g * 8]);
    v16h ap1 = ldfrag(&smem[PS_OFF + h * PS_HEAD + ((qt0 + 1) * 16 + l16) * PS_STR + kk * 32 + g * 8]);
    oacc[0][0] = wmma_f16(ap0, bv0, oacc[0][0]);
    oacc[0][1] = wmma_f16(ap0, bv1, oacc[0][1]);
    oacc[1][0] = wmma_f16(ap1, bv0, oacc[1][0]);
    oacc[1][1] = wmma_f16(ap1, bv1, oacc[1][1]);
  }

  // store: out[b][n][h*32 + d], coalesced over l16
#pragma unroll
  for (int qi = 0; qi < 2; ++qi) {
#pragma unroll
    for (int dt = 0; dt < 2; ++dt) {
#pragma unroll
      for (int r = 0; r < 8; ++r) {
        int n = (qt0 + qi) * 16 + g * 8 + r;
        if (n < NTOK) {
          int c = h * HD + dt * 16 + l16;
          out[((size_t)b * NTOK + n) * DIM + c] = oacc[qi][dt][r];
        }
      }
    }
  }
}

extern "C" void kernel_launch(void* const* d_in, const int* in_sizes, int n_in,
                              void* d_out, int out_size, void* d_ws, size_t ws_size,
                              hipStream_t stream) {
  const float* x         = (const float*)d_in[0];
  const float* mask      = (const float*)d_in[1];
  const float* qkv_w     = (const float*)d_in[2];
  const float* qkv_b     = (const float*)d_in[3];
  const float* rel_table = (const float*)d_in[4];
  float* out = (float*)d_out;

  _Float16* w16 = (_Float16*)d_ws;   // 384*128*2 = 96 KB scratch
  const int wn = 3 * DIM * DIM;      // 49152
  cvt_w_f16<<<(wn + 255) / 256, 256, 0, stream>>>(qkv_w, w16, wn);

  swin_attn_kernel<<<4096, 256, 0, stream>>>(x, mask, qkv_b, rel_table, w16, out);
}
